// CausalMultiheadAttention_17867063951442
// MI455X (gfx1250) — compile-verified, matrix operand order checked by reading, not run on hardware
//
#include <hip/hip_runtime.h>
#include <hip/hip_bf16.h>

// ---------------------------------------------------------------------------
// CDNA5 (gfx1250) causal multi-head attention, bf16 WMMA path.
//   x[B,S,2048]f32 --gemm--> qkv[B,S,6144]bf16 --rope(bf16)--> flash attn
//   --> attn[B,S,2048]f32 --gemm--> out[B,S,2048]f32
// Matrix math: V_WMMA_F32_16X16X32_BF16 (f32 accumulate).
// qkv kept in bf16 to halve attention traffic and make K-tile staging a pure
// global_load_b128 -> ds_store_b128 copy.
// ---------------------------------------------------------------------------

typedef __attribute__((ext_vector_type(16))) __bf16        v16bf;
typedef __attribute__((ext_vector_type(8)))  __bf16        bf16x8;
typedef __attribute__((ext_vector_type(8)))  float         v8f;
typedef __attribute__((ext_vector_type(4)))  float         f32x4;
typedef __attribute__((ext_vector_type(2)))  unsigned int  u32x2;

#define D_MODEL   2048
#define NUM_HEADS 16
#define HEAD_DIM  128
#define E_QKV     6144

// f32 -> bf16 (RNE), scalar.
__device__ __forceinline__ __bf16 f2bf(float f) {
    unsigned int u = __builtin_bit_cast(unsigned int, f);
    u += 0x7FFFu + ((u >> 16) & 1u);
    unsigned short h = (unsigned short)(u >> 16);
    return __builtin_bit_cast(__bf16, h);
}
// bf16 -> f32 (exact).
__device__ __forceinline__ float bf2f(__bf16 h) {
    unsigned int u = (unsigned int)__builtin_bit_cast(unsigned short, h) << 16;
    return __builtin_bit_cast(float, u);
}
// Pack two f32 into a packed-bf16 dword (a -> low, b -> high) with v_perm_b32.
__device__ __forceinline__ unsigned int pack2bf(float a, float b) {
    unsigned int ua = __builtin_bit_cast(unsigned int, a);
    unsigned int ub = __builtin_bit_cast(unsigned int, b);
    ua += 0x7FFFu + ((ua >> 16) & 1u);
    ub += 0x7FFFu + ((ub >> 16) & 1u);
    return __builtin_amdgcn_perm(ub, ua, 0x07060302u);
}

// K-index striping shared by 16-bit A (16x32) and B (32x16) fragments:
// lanes 0-15 hold K {0..7,16..23}, lanes 16-31 hold K {8..15,24..31}.
__device__ __forceinline__ int kmap(int j, int half) {
    return (j < 8) ? (half * 8 + j) : (16 + half * 8 + (j - 8));
}

__device__ __forceinline__ void store_out(float* p, float v)  { *p = v; }
__device__ __forceinline__ void store_out(__bf16* p, float v) { *p = f2bf(v); }

// ---------------------------------------------------------------------------
// C[M,N] (OutT) = A[M,K] (f32) * W[N,K]^T (f32), computed in bf16 WMMA.
// Block: 256 threads (8 waves). Block tile 128x128, wave tile 64x32,
// K staged 64 at a time (2 wmma-chunks per stage).
// ---------------------------------------------------------------------------
#define GSTR 72            // LDS row stride in bf16 (64 + 8 pad)
template <typename OutT>
__global__ __launch_bounds__(256)
void gemm_nt_bf16(const float* __restrict__ A, const float* __restrict__ W,
                  OutT* __restrict__ C, int M, int N, int K) {
    __shared__ __bf16 ldsA[128 * GSTR];
    __shared__ __bf16 ldsB[128 * GSTR];
    unsigned int* ldsAu = (unsigned int*)ldsA;
    unsigned int* ldsBu = (unsigned int*)ldsB;

    const int tid  = threadIdx.x;
    const int lane = tid & 31;
    const int wave = tid >> 5;          // 0..7
    const int ln   = lane & 15;
    const int half = (lane >> 4) & 1;
    const int wm   = wave >> 2;         // 0..1  (64 rows each)
    const int wn   = wave & 3;          // 0..3  (32 cols each)
    const int m0   = blockIdx.y * 128;
    const int n0   = blockIdx.x * 128;
    const int ldk4 = K >> 2;            // row stride in float4

    const v8f vz = {0.f,0.f,0.f,0.f,0.f,0.f,0.f,0.f};
    v8f acc[4][2];
#pragma unroll
    for (int fa = 0; fa < 4; ++fa)
#pragma unroll
        for (int fb = 0; fb < 2; ++fb) acc[fa][fb] = vz;

    const f32x4* A4 = (const f32x4*)A + (size_t)m0 * ldk4;
    const f32x4* W4 = (const f32x4*)W + (size_t)n0 * ldk4;

    for (int k0 = 0; k0 < K; k0 += 64) {
        __syncthreads();
        // Stage 128x64 tiles of A and W: float4 loads, packed-bf16 b64 stores.
#pragma unroll
        for (int i = 0; i < 8; ++i) {
            int idx = i * 256 + tid;       // 0..2047 float4 slots
            int r   = idx >> 4;            // 16 float4 per row
            int c4  = idx & 15;
            f32x4 av = A4[(size_t)r * ldk4 + (k0 >> 2) + c4];
            f32x4 wv = W4[(size_t)r * ldk4 + (k0 >> 2) + c4];
            u32x2 ap, wp;
            ap.x = pack2bf(av.x, av.y); ap.y = pack2bf(av.z, av.w);
            wp.x = pack2bf(wv.x, wv.y); wp.y = pack2bf(wv.z, wv.w);
            *(u32x2*)&ldsAu[r * (GSTR / 2) + c4 * 2] = ap;
            *(u32x2*)&ldsBu[r * (GSTR / 2) + c4 * 2] = wp;
        }
        if (k0 + 64 < K) {  // speculative prefetch of next K tile
            __builtin_prefetch(A + (size_t)(m0 + (tid & 127)) * K + k0 + 64, 0, 1);
            __builtin_prefetch(W + (size_t)(n0 + (tid & 127)) * K + k0 + 64, 0, 1);
        }
        __syncthreads();

#pragma unroll
        for (int cc = 0; cc < 2; ++cc) {
            v16bf af[4], bfr[2];
#pragma unroll
            for (int fa = 0; fa < 4; ++fa)
#pragma unroll
                for (int j = 0; j < 16; ++j)
                    af[fa][j] = ldsA[(wm * 64 + fa * 16 + ln) * GSTR + cc * 32 + kmap(j, half)];
#pragma unroll
            for (int fb = 0; fb < 2; ++fb)
#pragma unroll
                for (int j = 0; j < 16; ++j)
                    bfr[fb][j] = ldsB[(wn * 32 + fb * 16 + ln) * GSTR + cc * 32 + kmap(j, half)];

#pragma unroll
            for (int fa = 0; fa < 4; ++fa)
#pragma unroll
                for (int fb = 0; fb < 2; ++fb)
                    acc[fa][fb] = __builtin_amdgcn_wmma_f32_16x16x32_bf16(
                        false, af[fa], false, bfr[fb], (short)0, acc[fa][fb], false, false);
        }
    }

    // Epilogue: C/D layout row = e + 8*half, col = ln within each 16x16 tile.
#pragma unroll
    for (int fa = 0; fa < 4; ++fa)
#pragma unroll
        for (int fb = 0; fb < 2; ++fb)
#pragma unroll
            for (int e = 0; e < 8; ++e) {
                int row = m0 + wm * 64 + fa * 16 + 8 * half + e;
                int col = n0 + wn * 32 + fb * 16 + ln;
                store_out(&C[(size_t)row * N + col], acc[fa][fb][e]);
            }
}

// ---------------------------------------------------------------------------
// RoPE applied in-place to q and k slices of bf16 qkv[B,S,6144].
// One thread per (b,s,h,pair): dword load, rotate in f32, packed dword store.
// ---------------------------------------------------------------------------
__global__ __launch_bounds__(256)
void rope_kernel(__bf16* __restrict__ qkv, const int* __restrict__ pos, int B, int S) {
    int tid = blockIdx.x * blockDim.x + threadIdx.x;
    int i = tid & 63;            // pair index within head (dim_id = 2i)
    int t = tid >> 6;
    int h = t & 15;
    t >>= 4;
    int s = t % S;
    int b = t / S;
    if (b >= B) return;

    float p    = (float)pos[s];
    float freq = __expf(-((float)(2 * i) * (1.0f / (float)HEAD_DIM)) * __logf(10000.0f));
    float ang  = p * freq;
    float sn, cs;
    __sincosf(ang, &sn, &cs);

    size_t base = ((size_t)(b * S + s)) * E_QKV + (size_t)h * HEAD_DIM + 2 * i;
    unsigned int* qp = (unsigned int*)&qkv[base];            // 4B aligned (2i even)
    unsigned int* kp = (unsigned int*)&qkv[base + D_MODEL];
    unsigned int qv = *qp, kv = *kp;
    float q0 = __builtin_bit_cast(float, qv << 16);
    float q1 = __builtin_bit_cast(float, qv & 0xFFFF0000u);
    float k0 = __builtin_bit_cast(float, kv << 16);
    float k1 = __builtin_bit_cast(float, kv & 0xFFFF0000u);
    *qp = pack2bf(cs * q0 - sn * q1, sn * q0 + cs * q1);
    *kp = pack2bf(cs * k0 - sn * k1, sn * k0 + cs * k1);
}

// ---------------------------------------------------------------------------
// Flash attention: grid (S/128, H, B), 256 threads (8 waves).
// Each wave owns 16 q rows; block streams causal K/V in 32-row bf16 tiles.
// ---------------------------------------------------------------------------
#define KSTR 136           // ldsK row stride (128 + 8)
#define VSTR 40            // ldsVt row stride (32 + 8)
__global__ __launch_bounds__(256)
void flash_attn_kernel(const __bf16* __restrict__ qkv, float* __restrict__ out,
                       int B, int S) {
    __shared__ __bf16 ldsK [32 * KSTR];       // [kpos][d]
    __shared__ __bf16 ldsVt[128 * VSTR];      // [d][kpos] transposed
    __shared__ __bf16 ldsP [8 * 16 * 40];     // per-wave P re-marshal patch

    const int tid  = threadIdx.x;
    const int lane = tid & 31;
    const int wave = tid >> 5;                // 0..7
    const int ln   = lane & 15;
    const int half = (lane >> 4) & 1;
    const int q0   = blockIdx.x * 128;
    const int h    = blockIdx.y;
    const int b    = blockIdx.z;
    const float scale = 0.08838834764831845f; // 1/sqrt(128)

    const __bf16* base = qkv + ((size_t)(b * S)) * E_QKV + (size_t)h * HEAD_DIM;

    // Q fragments (16 rows x 128, A-layout), raw bf16, vectorized b128 loads.
    v16bf qa[4];
    {
        const bf16x8* qp8 =
            (const bf16x8*)(base + (size_t)(q0 + wave * 16 + ln) * E_QKV);
#pragma unroll
        for (int c = 0; c < 4; ++c) {
            bf16x8 lo = qp8[c * 4 + half];        // K = c*32 + half*8 ..+7
            bf16x8 hi = qp8[c * 4 + 2 + half];    // K = c*32 + 16 + half*8 ..+7
#pragma unroll
            for (int j = 0; j < 8; ++j) { qa[c][j] = lo[j]; qa[c][8 + j] = hi[j]; }
        }
    }

    const v8f vz = {0.f,0.f,0.f,0.f,0.f,0.f,0.f,0.f};
    v8f acc[8];                               // O: 16 x 128 f32
#pragma unroll
    for (int t = 0; t < 8; ++t) acc[t] = vz;
    float mrow[8], lrow[8];                   // row = e + 8*half
#pragma unroll
    for (int e = 0; e < 8; ++e) { mrow[e] = -1e30f; lrow[e] = 0.f; }

    const int nkb = q0 / 32 + 4;              // k tiles 0 .. q0+127
    for (int kb = 0; kb < nkb; ++kb) {
        __syncthreads();
        // Stage K (b128 copy) and V (transposed b16 scatter), both bf16.
        const bf16x8* kg8 = (const bf16x8*)(base + (size_t)(kb * 32) * E_QKV + D_MODEL);
        const bf16x8* vg8 = (const bf16x8*)(base + (size_t)(kb * 32) * E_QKV + 2 * D_MODEL);
#pragma unroll
        for (int i = 0; i < 2; ++i) {
            int idx = i * 256 + tid;          // 0..511 8-elem slots
            int r   = idx >> 4;               // 16 slots per 128-wide row
            int c8  = idx & 15;
            int d0  = c8 * 8;
            bf16x8 kv = kg8[(size_t)r * (E_QKV / 8) + c8];
            bf16x8 vv = vg8[(size_t)r * (E_QKV / 8) + c8];
            *(bf16x8*)&ldsK[r * KSTR + d0] = kv;
#pragma unroll
            for (int m = 0; m < 8; ++m)
                ldsVt[(d0 + m) * VSTR + r] = vv[m];
        }
        __syncthreads();

        // S = Q * K^T  (16 x 32 scores as two 16x16 f32 fragments)
        v8f sc[2];
        sc[0] = vz; sc[1] = vz;
#pragma unroll
        for (int t = 0; t < 2; ++t)
#pragma unroll
            for (int c = 0; c < 4; ++c) {
                v16bf kf;
#pragma unroll
                for (int j = 0; j < 16; ++j)
                    kf[j] = ldsK[(t * 16 + ln) * KSTR + c * 32 + kmap(j, half)];
                sc[t] = __builtin_amdgcn_wmma_f32_16x16x32_bf16(
                    false, qa[c], false, kf, (short)0, sc[t], false, false);
            }

        // Scale + causal mask + online softmax (row reductions over 16 lanes).
        const int rbase = q0 + wave * 16 + 8 * half;
#pragma unroll
        for (int e = 0; e < 8; ++e) {
            int rg = rbase + e;
            int c0 = kb * 32 + ln;
            float s0 = sc[0][e] * scale;
            float s1 = sc[1][e] * scale;
            if (c0      > rg) s0 = -1e30f;
            if (c0 + 16 > rg) s1 = -1e30f;

            float mx = fmaxf(s0, s1);
#pragma unroll
            for (int d = 1; d < 16; d <<= 1) mx = fmaxf(mx, __shfl_xor(mx, d, 16));
            float mnew = fmaxf(mrow[e], mx);

            float ex0 = __expf(s0 - mnew);
            float ex1 = __expf(s1 - mnew);
            sc[0][e] = ex0; sc[1][e] = ex1;
            float rsum = ex0 + ex1;
#pragma unroll
            for (int d = 1; d < 16; d <<= 1) rsum += __shfl_xor(rsum, d, 16);

            float corr = __expf(mrow[e] - mnew);
            lrow[e] = lrow[e] * corr + rsum;
            mrow[e] = mnew;
#pragma unroll
            for (int t = 0; t < 8; ++t) acc[t][e] *= corr;
        }

        // Re-marshal P (C-layout -> A-layout) through per-wave LDS patch.
#pragma unroll
        for (int t = 0; t < 2; ++t)
#pragma unroll
            for (int e = 0; e < 8; ++e)
                ldsP[wave * 640 + (e + 8 * half) * 40 + t * 16 + ln] = f2bf(sc[t][e]);
        __builtin_amdgcn_wave_barrier();      // DS is in-order per wave
        v16bf pa;
#pragma unroll
        for (int j = 0; j < 16; ++j)
            pa[j] = ldsP[wave * 640 + ln * 40 + kmap(j, half)];

        // O += P * V  (eight 16-col sub-tiles)
#pragma unroll
        for (int t = 0; t < 8; ++t) {
            v16bf vf;
#pragma unroll
            for (int j = 0; j < 16; ++j)
                vf[j] = ldsVt[(t * 16 + ln) * VSTR + kmap(j, half)];
            acc[t] = __builtin_amdgcn_wmma_f32_16x16x32_bf16(
                false, pa, false, vf, (short)0, acc[t], false, false);
        }
    }

    // Finalize: O /= l, write heads-merged f32 [B,S,2048].
    float* ob = out + ((size_t)(b * S)) * D_MODEL + (size_t)h * HEAD_DIM;
#pragma unroll
    for (int e = 0; e < 8; ++e) {
        float inv = 1.0f / lrow[e];
        int rg = q0 + wave * 16 + 8 * half + e;
#pragma unroll
        for (int t = 0; t < 8; ++t)
            ob[(size_t)rg * D_MODEL + t * 16 + ln] = acc[t][e] * inv;
    }
}

// ---------------------------------------------------------------------------
extern "C" void kernel_launch(void* const* d_in, const int* in_sizes, int n_in,
                              void* d_out, int out_size, void* d_ws, size_t ws_size,
                              hipStream_t stream) {
    const float* x     = (const float*)d_in[0];
    const int*   pos   = (const int*)  d_in[1];
    const float* qkv_w = (const float*)d_in[2];
    const float* o_w   = (const float*)d_in[3];
    float*       out   = (float*)d_out;

    const int B = 2, S = 2048;
    const int M = B * S;                      // 4096

    __bf16* qkv  = (__bf16*)d_ws;                              // M x 6144 bf16 (~50 MB)
    float*  attn = (float*)((char*)d_ws + (size_t)M * E_QKV * 2); // M x 2048 f32 (~34 MB)

    // 1) qkv = bf16(x @ qkv_w^T)
    gemm_nt_bf16<__bf16><<<dim3(E_QKV / 128, M / 128), dim3(256), 0, stream>>>(
        x, qkv_w, qkv, M, E_QKV, D_MODEL);

    // 2) RoPE on q,k (in place, bf16)
    rope_kernel<<<(B * S * NUM_HEADS * (HEAD_DIM / 2)) / 256, 256, 0, stream>>>(
        qkv, pos, B, S);

    // 3) flash attention -> attn (f32)
    flash_attn_kernel<<<dim3(S / 128, NUM_HEADS, B), dim3(256), 0, stream>>>(
        qkv, attn, B, S);

    // 4) out = attn @ o_w^T (f32)
    gemm_nt_bf16<float><<<dim3(D_MODEL / 128, M / 128), dim3(256), 0, stream>>>(
        attn, o_w, out, M, D_MODEL, D_MODEL);
}
